// JustAttentionDropOutGCN_50130858279703
// MI455X (gfx1250) — matrix operand order repaired and checked by reading, not compiled, not run on hardware
//
#include <hip/hip_runtime.h>

// ---------------- problem constants ----------------
constexpr int kB   = 8;
constexpr int kN   = 128;
constexpr int kT   = 64;
constexpr int kIN  = 2;
constexpr int kH   = 128;
constexpr int kNH  = 8;
constexpr int kHD  = 16;   // kH / kNH
constexpr int kDFF = 512;
constexpr int kNL  = 5;
constexpr int kNG  = 5;
constexpr int kBN  = kB * kN;      // 1024
constexpr int kM   = kBN * kT;     // 65536 token rows

// ---------------- WMMA types / helpers ----------------
typedef __attribute__((ext_vector_type(16))) __bf16        v16bf;
typedef __attribute__((ext_vector_type(8)))  float         v8f;
typedef __attribute__((ext_vector_type(4)))  unsigned int  u32x4;

union FragBf {            // raw-bits view of a 16-element bf16 fragment
    v16bf v;
    u32x4 u[2];
};

__device__ __forceinline__ unsigned short f32_to_bf16_bits(float f) {
    unsigned u = __builtin_bit_cast(unsigned, f);
    unsigned r = u + 0x7fffu + ((u >> 16) & 1u);   // round-to-nearest-even
    return (unsigned short)(r >> 16);
}

// A fragment (16x32, k-contiguous): 16 consecutive ushorts
__device__ __forceinline__ v16bf load_afrag(const unsigned short* p) {
    FragBf f;
    f.u[0] = *(const u32x4*)(p);
    f.u[1] = *(const u32x4*)(p + 8);
    return f.v;
}
// B fragment (32x16, k-contiguous storage): 16 consecutive ushorts
__device__ __forceinline__ v16bf load_bfrag(const unsigned short* p) {
    FragBf f;
    f.u[0] = *(const u32x4*)(p);
    f.u[1] = *(const u32x4*)(p + 8);
    return f.v;
}

// ---------------- kernel 1: mask m[t, b*N+n] = ego[b,t,n] ----------------
__global__ void k_mask(const unsigned char* __restrict__ ego, float* __restrict__ mvec) {
    int tid = blockIdx.x * blockDim.x + threadIdx.x;
    if (tid >= kT * kBN) return;
    int t = tid / kBN, bn = tid % kBN;
    int b = bn / kN, n = bn % kN;
    mvec[tid] = ego[((size_t)b * kT + t) * kN + n] ? 1.f : 0.f;
}

// ------- kernel 2: deg_i = m_i*(sum_j A_ij m_j + 1); dinv = rsqrt ---------
__global__ void k_dinv(const float* __restrict__ Adj, const float* __restrict__ mvec,
                       float* __restrict__ dinv) {
    int tid = blockIdx.x * blockDim.x + threadIdx.x;
    if (tid >= kT * kBN) return;
    int t = tid / kBN, i = tid % kBN;
    const float* row  = Adj + ((size_t)t * kBN + i) * kBN;
    const float* mrow = mvec + (size_t)t * kBN;
    float s = 0.f;
    for (int j = 0; j < kBN; ++j) s += row[j] * mrow[j];
    float deg = mrow[i] * (s + 1.f);
    dinv[tid] = (deg > 0.f) ? rsqrtf(deg) : 0.f;
}

// ---- kernel 3: materialize D^-1/2 (A+I) D^-1/2 as bf16, once ------------
__global__ void k_build_norm(const float* __restrict__ Adj, const float* __restrict__ mvec,
                             const float* __restrict__ dinv, unsigned short* __restrict__ Nbf) {
    size_t tid = (size_t)blockIdx.x * blockDim.x + threadIdx.x;
    if (tid >= (size_t)kT * kBN * kBN) return;
    int j = (int)(tid & (kBN - 1));
    size_t r = tid >> 10;
    int i = (int)(r & (kBN - 1));
    int t = (int)(r >> 10);
    const float* mrow = mvec + (size_t)t * kBN;
    const float* drow = dinv + (size_t)t * kBN;
    float a = Adj[tid] * mrow[i] * mrow[j];
    if (i == j) a += mrow[i];
    a *= drow[i] * drow[j];
    Nbf[tid] = f32_to_bf16_bits(a);
}

// ---- kernel 4: one-time weight convert + transpose: Wt[n,k] = bf(W[k,n]) -
__global__ void k_convW(const float* __restrict__ W, unsigned short* __restrict__ Wt,
                        int K, int N) {
    int tid = blockIdx.x * blockDim.x + threadIdx.x;
    if (tid >= K * N) return;
    int k = tid / N, n = tid % N;
    Wt[(size_t)n * K + k] = f32_to_bf16_bits(W[tid]);
}

// ---- kernel 5: P^T = ((x*m) @ W1)^T, bf16, per-t transposed layout ------
__global__ void k_gcn_in(const float* __restrict__ X, const float* __restrict__ W1,
                         const float* __restrict__ mvec, unsigned short* __restrict__ PbfT) {
    int tid = blockIdx.x * blockDim.x + threadIdx.x;
    if (tid >= kM * kH) return;
    int row = tid / kH, c = tid % kH;
    int t = row / kBN, i = row % kBN;
    float v = X[(size_t)row * kIN + 0] * W1[c] + X[(size_t)row * kIN + 1] * W1[kH + c];
    PbfT[((size_t)t * kH + c) * kBN + i] = f32_to_bf16_bits(mvec[row] * v);
}

// ---- kernel 6: H = relu(Nrm_t @ P_t + bias), 2x2-blocked WMMA ----------
// Nbf: [T][BN][BN] bf16 (A).  PbfT: [T][H][BN] bf16 (B, k-contig).
// block = 8 waves as 2(M) x 4(N); each wave owns a 32x32 macro-tile.
// grid = (BN/64, T).
__global__ void k_norm_mm(const unsigned short* __restrict__ Nbf,
                          const unsigned short* __restrict__ PbfT,
                          const float* __restrict__ bias,
                          float* __restrict__ Hout,            // [T][BN][H] f32
                          unsigned short* __restrict__ HoutBf) // [T*BN][H] bf16
{
    const int t    = blockIdx.y;
    const int lane = threadIdx.x & 31;
    const int wv   = threadIdx.x >> 5;
    const int wm   = wv >> 2;           // 0..1
    const int wn   = wv & 3;            // 0..3
    const int rm0  = blockIdx.x * 64 + wm * 32;
    const int cn0  = wn * 32;
    const int mr = lane & 15;
    const int hi = lane >> 4;
    const unsigned short* a0 = Nbf + (size_t)t * kBN * kBN + (size_t)(rm0 + mr) * kBN + hi * 16;
    const unsigned short* a1 = a0 + (size_t)16 * kBN;
    const unsigned short* b0 = PbfT + (size_t)t * kH * kBN + (size_t)(cn0 + mr) * kBN + hi * 16;
    const unsigned short* b1 = b0 + (size_t)16 * kBN;
    v8f acc00 = {}, acc01 = {}, acc10 = {}, acc11 = {};
    for (int k0 = 0; k0 < kBN; k0 += 32) {
        v16bf fa0 = load_afrag(a0 + k0);
        v16bf fa1 = load_afrag(a1 + k0);
        v16bf fb0 = load_bfrag(b0 + k0);
        v16bf fb1 = load_bfrag(b1 + k0);
        if (k0 + 32 < kBN) __builtin_prefetch(a0 + k0 + 32, 0, 1);
        acc00 = __builtin_amdgcn_wmma_f32_16x16x32_bf16(false, fa0, false, fb0, (short)0, acc00, false, false);
        acc01 = __builtin_amdgcn_wmma_f32_16x16x32_bf16(false, fa0, false, fb1, (short)0, acc01, false, false);
        acc10 = __builtin_amdgcn_wmma_f32_16x16x32_bf16(false, fa1, false, fb0, (short)0, acc10, false, false);
        acc11 = __builtin_amdgcn_wmma_f32_16x16x32_bf16(false, fa1, false, fb1, (short)0, acc11, false, false);
    }
    #pragma unroll
    for (int i = 0; i < 2; ++i) {
        #pragma unroll
        for (int j = 0; j < 2; ++j) {
            const v8f* accp = (i == 0) ? (j == 0 ? &acc00 : &acc01)
                                       : (j == 0 ? &acc10 : &acc11);
            int gn = cn0 + j * 16 + mr;
            float bval = bias[gn];
            #pragma unroll
            for (int v = 0; v < 8; ++v) {
                int gm = rm0 + i * 16 + hi * 8 + v;
                float o = fmaxf((*accp)[v] + bval, 0.f);
                size_t idx = ((size_t)t * kBN + gm) * kH + gn;
                Hout[idx] = o;
                HoutBf[idx] = f32_to_bf16_bits(o);
            }
        }
    }
}

// ---- kernel 7: generic C = act(A @ W + bias), 2x2-blocked WMMA ----------
// A: [M x K] bf16 row-major.  Wt: [N x K] bf16 (pre-transposed, k-contig).
// block = 8 waves as 2(M) x 4(N): 64 rows x 128 cols; grid = (M/64, N/128).
// Optional outputs: Cf (f32 row-major), Cbf (bf16 row-major),
// CbfT (bf16 per-t transposed [T][N][BN] -- GCN P operand).
__global__ void k_gemm(const unsigned short* __restrict__ A,
                       const unsigned short* __restrict__ Wt,
                       const float* __restrict__ bias,
                       float* __restrict__ Cf,
                       unsigned short* __restrict__ Cbf,
                       unsigned short* __restrict__ CbfT,
                       int K, int N, int do_relu) {
    const int lane = threadIdx.x & 31;
    const int wv   = threadIdx.x >> 5;
    const int wm   = wv >> 2;
    const int wn   = wv & 3;
    const int rm0  = blockIdx.x * 64 + wm * 32;
    const int cn0  = blockIdx.y * 128 + wn * 32;
    const int mr = lane & 15;
    const int hi = lane >> 4;
    const unsigned short* a0 = A + (size_t)(rm0 + mr) * K + hi * 16;
    const unsigned short* a1 = a0 + (size_t)16 * K;
    const unsigned short* b0 = Wt + (size_t)(cn0 + mr) * K + hi * 16;
    const unsigned short* b1 = b0 + (size_t)16 * K;
    v8f acc00 = {}, acc01 = {}, acc10 = {}, acc11 = {};
    for (int k0 = 0; k0 < K; k0 += 32) {
        v16bf fa0 = load_afrag(a0 + k0);
        v16bf fa1 = load_afrag(a1 + k0);
        v16bf fb0 = load_bfrag(b0 + k0);
        v16bf fb1 = load_bfrag(b1 + k0);
        if (k0 + 32 < K) __builtin_prefetch(a0 + k0 + 32, 0, 1);
        acc00 = __builtin_amdgcn_wmma_f32_16x16x32_bf16(false, fa0, false, fb0, (short)0, acc00, false, false);
        acc01 = __builtin_amdgcn_wmma_f32_16x16x32_bf16(false, fa0, false, fb1, (short)0, acc01, false, false);
        acc10 = __builtin_amdgcn_wmma_f32_16x16x32_bf16(false, fa1, false, fb0, (short)0, acc10, false, false);
        acc11 = __builtin_amdgcn_wmma_f32_16x16x32_bf16(false, fa1, false, fb1, (short)0, acc11, false, false);
    }
    #pragma unroll
    for (int i = 0; i < 2; ++i) {
        #pragma unroll
        for (int j = 0; j < 2; ++j) {
            const v8f* accp = (i == 0) ? (j == 0 ? &acc00 : &acc01)
                                       : (j == 0 ? &acc10 : &acc11);
            const int gn  = cn0 + j * 16 + mr;
            const int gm0 = rm0 + i * 16 + hi * 8;
            const float bval = bias ? bias[gn] : 0.f;
            float o[8];
            #pragma unroll
            for (int v = 0; v < 8; ++v) {
                o[v] = (*accp)[v] + bval;
                if (do_relu) o[v] = fmaxf(o[v], 0.f);
            }
            if (Cf) {
                #pragma unroll
                for (int v = 0; v < 8; ++v) Cf[(size_t)(gm0 + v) * N + gn] = o[v];
            }
            if (Cbf) {
                #pragma unroll
                for (int v = 0; v < 8; ++v)
                    Cbf[(size_t)(gm0 + v) * N + gn] = f32_to_bf16_bits(o[v]);
            }
            if (CbfT) {   // per-t transposed: 8 consecutive ushorts -> one b128
                int tt = gm0 / kBN;
                int i0 = gm0 % kBN;
                u32x4 pk;
                #pragma unroll
                for (int p = 0; p < 4; ++p)
                    pk[p] = (unsigned)f32_to_bf16_bits(o[2 * p]) |
                            ((unsigned)f32_to_bf16_bits(o[2 * p + 1]) << 16);
                *(u32x4*)(CbfT + ((size_t)tt * kH + gn) * kBN + i0) = pk;
            }
        }
    }
}

// ---- kernel 8: x[bn,t,:] = h[t,bn,:] + pos-embed (f32 + bf16 copies) ----
__global__ void k_build_x(const float* __restrict__ Hin, float* __restrict__ X,
                          unsigned short* __restrict__ Xbf) {
    int tid = blockIdx.x * blockDim.x + threadIdx.x;
    if (tid >= kM * kH) return;
    int h = tid % kH;
    int tmp = tid / kH;
    int t = tmp % kT;
    int bn = tmp / kT;
    int j = h >> 1;
    float div = expf(-logf(10000.f) * (float)(2 * j) / (float)kH);
    float ang = (float)t * div;
    float pe = (h & 1) ? cosf(ang) : sinf(ang);
    float o = Hin[((size_t)t * kBN + bn) * kH + h] + pe;
    X[tid] = o;
    Xbf[tid] = f32_to_bf16_bits(o);
}

// ---- kernel 9: attention per (node, head); output straight to bf16 ------
__global__ void k_attn(const float* __restrict__ Q, const float* __restrict__ Kk,
                       const float* __restrict__ V, unsigned short* __restrict__ Obf) {
    const int bn = blockIdx.x;
    const int hd = blockIdx.y;
    const int qr = threadIdx.x;                 // query row 0..63
    const size_t base = (size_t)bn * kT * kH + (size_t)hd * kHD;
    float qv[kHD];
    #pragma unroll
    for (int d = 0; d < kHD; ++d) qv[d] = Q[base + (size_t)qr * kH + d];
    float s[kT];
    float mx = -1e30f;
    for (int j = 0; j < kT; ++j) {
        const float* kp = Kk + base + (size_t)j * kH;
        float a = 0.f;
        #pragma unroll
        for (int d = 0; d < kHD; ++d) a += qv[d] * kp[d];
        a *= 0.25f;                             // 1/sqrt(HD)
        s[j] = a;
        mx = fmaxf(mx, a);
    }
    float sum = 0.f;
    for (int j = 0; j < kT; ++j) { s[j] = expf(s[j] - mx); sum += s[j]; }
    float inv = 1.f / sum;
    float o[kHD] = {};
    for (int j = 0; j < kT; ++j) {
        float p = s[j] * inv;
        const float* vp = V + base + (size_t)j * kH;
        #pragma unroll
        for (int d = 0; d < kHD; ++d) o[d] += p * vp[d];
    }
    #pragma unroll
    for (int d = 0; d < kHD; ++d)
        Obf[base + (size_t)qr * kH + d] = f32_to_bf16_bits(o[d]);
}

// ---- kernel 10: x = LN(x + y); also refresh bf16 copy of x --------------
__global__ void k_res_ln(float* __restrict__ X, const float* __restrict__ Y,
                         const float* __restrict__ g, const float* __restrict__ b,
                         unsigned short* __restrict__ Xbf) {
    __shared__ float red[kH];
    const int row = blockIdx.x;
    const int c = threadIdx.x;
    float s = X[(size_t)row * kH + c] + Y[(size_t)row * kH + c];
    red[c] = s;
    __syncthreads();
    for (int off = kH / 2; off > 0; off >>= 1) {
        if (c < off) red[c] += red[c + off];
        __syncthreads();
    }
    float mean = red[0] * (1.f / kH);
    __syncthreads();
    float d = s - mean;
    red[c] = d * d;
    __syncthreads();
    for (int off = kH / 2; off > 0; off >>= 1) {
        if (c < off) red[c] += red[c + off];
        __syncthreads();
    }
    float var = red[0] * (1.f / kH);
    float o = d * rsqrtf(var + 1e-5f) * g[c] + b[c];
    X[(size_t)row * kH + c] = o;
    Xbf[(size_t)row * kH + c] = f32_to_bf16_bits(o);
}

// ---------------------------- launcher -----------------------------------
extern "C" void kernel_launch(void* const* d_in, const int* in_sizes, int n_in,
                              void* d_out, int out_size, void* d_ws, size_t ws_size,
                              hipStream_t stream) {
    const unsigned char* ego = (const unsigned char*)d_in[0];
    const float* Xpos = (const float*)d_in[1];
    const float* Adj  = (const float*)d_in[2];
    const float* gw1  = (const float*)d_in[3];
    const float* gb1  = (const float*)d_in[4];
    const float* gw   = (const float*)d_in[5];
    const float* gb   = (const float*)d_in[6];
    const float* wq   = (const float*)d_in[7];
    const float* wk   = (const float*)d_in[8];
    const float* wv   = (const float*)d_in[9];
    const float* wo   = (const float*)d_in[10];
    const float* bq   = (const float*)d_in[11];
    const float* bk   = (const float*)d_in[12];
    const float* bv   = (const float*)d_in[13];
    const float* bo   = (const float*)d_in[14];
    const float* l1g  = (const float*)d_in[15];
    const float* l1b  = (const float*)d_in[16];
    const float* l2g  = (const float*)d_in[17];
    const float* l2b  = (const float*)d_in[18];
    const float* fw1  = (const float*)d_in[19];
    const float* fb1  = (const float*)d_in[20];
    const float* fw2  = (const float*)d_in[21];
    const float* fb2  = (const float*)d_in[22];

    // ---- workspace carve (256B aligned chunks) ----
    char* p = (char*)d_ws;
    auto carve = [&](size_t bytes) -> void* {
        void* r = (void*)p;
        p += (bytes + 255) & ~(size_t)255;
        return r;
    };
    unsigned short* Nbf  = (unsigned short*)carve((size_t)kT * kBN * kBN * 2); // 134 MB
    float* mvec = (float*)carve((size_t)kT * kBN * 4);
    float* dinv = (float*)carve((size_t)kT * kBN * 4);
    float* hA   = (float*)carve((size_t)kM * kH * 4);   // GCN h (f32)
    float* hB   = (float*)carve((size_t)kM * kH * 4);   // residual branch (f32)
    float* qb   = (float*)carve((size_t)kM * kH * 4);
    float* kb   = (float*)carve((size_t)kM * kH * 4);
    float* vb   = (float*)carve((size_t)kM * kH * 4);
    unsigned short* hAbf = (unsigned short*)carve((size_t)kM * kH * 2); // h bf16 / attn-out bf16
    unsigned short* PbfT = (unsigned short*)carve((size_t)kM * kH * 2); // P^T bf16 / FFN hidden bf16
    unsigned short* xbf  = (unsigned short*)carve((size_t)kM * kH * 2);
    // bf16 transposed weights
    unsigned short* gwT  = (unsigned short*)carve((size_t)kNG * kH * kH * 2);
    unsigned short* wqT  = (unsigned short*)carve((size_t)kNL * kH * kH * 2);
    unsigned short* wkT  = (unsigned short*)carve((size_t)kNL * kH * kH * 2);
    unsigned short* wvT  = (unsigned short*)carve((size_t)kNL * kH * kH * 2);
    unsigned short* woT  = (unsigned short*)carve((size_t)kNL * kH * kH * 2);
    unsigned short* fw1T = (unsigned short*)carve((size_t)kNL * kH * kDFF * 2);
    unsigned short* fw2T = (unsigned short*)carve((size_t)kNL * kDFF * kH * 2);
    float* x = (float*)d_out;                        // [BN,T,H] == [B,N,T,H] flat

    // ---- one-time weight convert+transpose ----
    const int cw = (kH * kH + 255) / 256;
    for (int i = 0; i < kNG; ++i)
        k_convW<<<cw, 256, 0, stream>>>(gw + (size_t)i * kH * kH, gwT + (size_t)i * kH * kH, kH, kH);
    for (int i = 0; i < kNL; ++i) {
        k_convW<<<cw, 256, 0, stream>>>(wq + (size_t)i * kH * kH, wqT + (size_t)i * kH * kH, kH, kH);
        k_convW<<<cw, 256, 0, stream>>>(wk + (size_t)i * kH * kH, wkT + (size_t)i * kH * kH, kH, kH);
        k_convW<<<cw, 256, 0, stream>>>(wv + (size_t)i * kH * kH, wvT + (size_t)i * kH * kH, kH, kH);
        k_convW<<<cw, 256, 0, stream>>>(wo + (size_t)i * kH * kH, woT + (size_t)i * kH * kH, kH, kH);
        k_convW<<<(kH * kDFF + 255) / 256, 256, 0, stream>>>(
            fw1 + (size_t)i * kH * kDFF, fw1T + (size_t)i * kH * kDFF, kH, kDFF);
        k_convW<<<(kDFF * kH + 255) / 256, 256, 0, stream>>>(
            fw2 + (size_t)i * kDFF * kH, fw2T + (size_t)i * kDFF * kH, kDFF, kH);
    }

    // ---- GCN ----
    k_mask<<<(kT * kBN) / 256, 256, 0, stream>>>(ego, mvec);
    k_dinv<<<(kT * kBN) / 256, 256, 0, stream>>>(Adj, mvec, dinv);
    k_build_norm<<<(unsigned)(((size_t)kT * kBN * kBN) / 256), 256, 0, stream>>>(Adj, mvec, dinv, Nbf);
    k_gcn_in<<<(kM * kH) / 256, 256, 0, stream>>>(Xpos, gw1, mvec, PbfT);
    dim3 gnm(kBN / 64, kT);
    k_norm_mm<<<gnm, 256, 0, stream>>>(Nbf, PbfT, gb1, hA, hAbf);
    for (int i = 0; i < kNG; ++i) {
        k_gemm<<<dim3(kM / 64, 1), 256, 0, stream>>>(
            hAbf, gwT + (size_t)i * kH * kH, nullptr,
            nullptr, nullptr, PbfT, kH, kH, 0);
        k_norm_mm<<<gnm, 256, 0, stream>>>(Nbf, PbfT, gb + (size_t)i * kH, hA, hAbf);
    }

    // ---- positional embedding / transpose into x ----
    k_build_x<<<(kM * kH) / 256, 256, 0, stream>>>(hA, x, xbf);

    // ---- transformer layers ----
    unsigned short* aobf = hAbf;   // reuse (GCN done)
    unsigned short* ffh  = PbfT;   // reuse (GCN done)
    for (int i = 0; i < kNL; ++i) {
        k_gemm<<<dim3(kM / 64, 1), 256, 0, stream>>>(
            xbf, wqT + (size_t)i * kH * kH, bq + (size_t)i * kH, qb, nullptr, nullptr, kH, kH, 0);
        k_gemm<<<dim3(kM / 64, 1), 256, 0, stream>>>(
            xbf, wkT + (size_t)i * kH * kH, bk + (size_t)i * kH, kb, nullptr, nullptr, kH, kH, 0);
        k_gemm<<<dim3(kM / 64, 1), 256, 0, stream>>>(
            xbf, wvT + (size_t)i * kH * kH, bv + (size_t)i * kH, vb, nullptr, nullptr, kH, kH, 0);
        k_attn<<<dim3(kBN, kNH), kT, 0, stream>>>(qb, kb, vb, aobf);
        k_gemm<<<dim3(kM / 64, 1), 256, 0, stream>>>(
            aobf, woT + (size_t)i * kH * kH, bo + (size_t)i * kH, hB, nullptr, nullptr, kH, kH, 0);
        k_res_ln<<<kM, kH, 0, stream>>>(x, hB, l1g + (size_t)i * kH, l1b + (size_t)i * kH, xbf);
        // FFN chunked 4x so the [M,512] bf16 hidden reuses the free P^T buffer
        const int rows = kM / 4;
        for (int c = 0; c < 4; ++c) {
            unsigned short* xc = xbf + (size_t)c * rows * kH;
            float* hBc = hB + (size_t)c * rows * kH;
            k_gemm<<<dim3(rows / 64, kDFF / 128), 256, 0, stream>>>(
                xc, fw1T + (size_t)i * kH * kDFF, fb1 + (size_t)i * kDFF,
                nullptr, ffh, nullptr, kH, kDFF, 1);
            k_gemm<<<dim3(rows / 64, 1), 256, 0, stream>>>(
                ffh, fw2T + (size_t)i * kDFF * kH, fb2 + (size_t)i * kH,
                hBc, nullptr, nullptr, kDFF, kH, 0);
        }
        k_res_ln<<<kM, kH, 0, stream>>>(x, hB, l2g + (size_t)i * kH, l2b + (size_t)i * kH, xbf);
    }
    (void)in_sizes; (void)n_in; (void)out_size; (void)ws_size;
}
